// MessagePassingModelDEBUG_49692771615269
// MI455X (gfx1250) — matrix operand order repaired and checked by reading, not compiled, not run on hardware
//
#include <hip/hip_runtime.h>
#include <hip/hip_bf16.h>
#include <math.h>

#define F 32

typedef __attribute__((ext_vector_type(16))) _Float16 v16h;
typedef __attribute__((ext_vector_type(8)))  float    v8f;

// ---------------------------------------------------------------------------
// Gaunt coefficients via the reference's quadrature (8-pt Gauss-Legendre x 16 phi)
// ---------------------------------------------------------------------------
__global__ void gaunt_kernel(float* __restrict__ G) {
  int i = threadIdx.x;
  if (i >= 729) return;
  int p = i / 81, q = (i / 9) % 9, r = i % 9;
  const double gt[8] = {-0.9602898564975363, -0.7966664774136267, -0.5255324099163290,
                        -0.1834346424956498,  0.1834346424956498,  0.5255324099163290,
                         0.7966664774136267,  0.9602898564975363};
  const double gw[8] = { 0.1012285362903763,  0.2223810344533745,  0.3137066458778873,
                         0.3626837833783620,  0.3626837833783620,  0.3137066458778873,
                         0.2223810344533745,  0.1012285362903763};
  const double TWO_PI_16 = 6.283185307179586476925286766559 / 16.0;
  double acc = 0.0;
  for (int it = 0; it < 8; ++it) {
    double t  = gt[it];
    double w  = gw[it] / 32.0;
    double st = sqrt(1.0 - t * t);
    for (int j = 0; j < 16; ++j) {
      double phi = j * TWO_PI_16;
      double x = st * cos(phi), y = st * sin(phi), z = t;
      double s3 = 1.7320508075688772935274463415059;
      double Yv[9] = {1.0, x, y, z, s3 * x * y, s3 * y * z, 0.5 * (3.0 * z * z - 1.0),
                      s3 * x * z, 0.5 * s3 * (x * x - y * y)};
      acc += w * Yv[p] * Yv[q] * Yv[r];
    }
  }
  G[i] = (fabs(acc) < 1e-10) ? 0.0f : (float)acc;
}

// ---------------------------------------------------------------------------
// Edge basis: spherical harmonics Y[E,9] and cutoff-weighted Bernstein rbf[E,8].
// Block 0 also warms L2 with the iter-0 message weights (global_prefetch_b8).
// ---------------------------------------------------------------------------
__global__ void edge_basis_kernel(const float* __restrict__ pos,
                                  const int* __restrict__ dst, const int* __restrict__ src,
                                  const float* __restrict__ mpW0,
                                  float* __restrict__ Yb, float* __restrict__ RBFb, int E) {
  if (blockIdx.x == 0) {
    // 256 threads x stride-27 covers the 27*256-float weight block (one line each)
    __builtin_prefetch(mpW0 + (size_t)threadIdx.x * 27, 0, 1);
  }
  int e = blockIdx.x * blockDim.x + threadIdx.x;
  if (e >= E) return;
  int s = src[e], d = dst[e];
  float dx = pos[s * 3 + 0] - pos[d * 3 + 0];
  float dy = pos[s * 3 + 1] - pos[d * 3 + 1];
  float dz = pos[s * 3 + 2] - pos[d * 3 + 2];
  float r2  = dx * dx + dy * dy + dz * dz;
  float r   = sqrtf(fmaxf(r2, 1e-12f));
  float inv = 1.0f / r;
  float x = dx * inv, y = dy * inv, z = dz * inv;
  const float s3 = 1.7320508075688772f;
  float Y[9] = {1.0f, x, y, z, s3 * x * y, s3 * y * z, 0.5f * (3.0f * z * z - 1.0f),
                s3 * x * z, 0.5f * s3 * (x * x - y * y)};
#pragma unroll
  for (int q = 0; q < 9; ++q) Yb[e * 9 + q] = Y[q];

  const float C = 5.0f;
  float rc  = fminf(r, C - 1e-6f);
  float cut = (r < C) ? expf(-(rc * rc) / ((C - rc) * (C + rc))) : 0.0f;
  float yv = 1.0f / (1.0f + r);
  float om = 1.0f - yv;
  float yp[8], mp[8];
  yp[0] = 1.0f; mp[0] = 1.0f;
#pragma unroll
  for (int k = 1; k < 8; ++k) { yp[k] = yp[k - 1] * yv; mp[k] = mp[k - 1] * om; }
  const float binom[8] = {1, 7, 21, 35, 35, 21, 7, 1};
#pragma unroll
  for (int k = 0; k < 8; ++k) RBFb[e * 8 + k] = binom[k] * yp[k] * mp[7 - k] * cut;
}

// ---------------------------------------------------------------------------
// Node embed init: x[:,0,:] = embed[Z], x[:,1:,:] = 0
// ---------------------------------------------------------------------------
__global__ void node_init_kernel(const float* __restrict__ embed, const int* __restrict__ Z,
                                 float* __restrict__ x, int N) {
  int i = blockIdx.x * blockDim.x + threadIdx.x;
  if (i >= N * F) return;
  int n = i >> 5, f = i & 31;
  float* b = x + n * 9 * F + f;
  b[0] = embed[Z[n] * F + f];
#pragma unroll
  for (int p = 1; p < 9; ++p) b[p * F] = 0.0f;
}

// ---------------------------------------------------------------------------
// Message passing: one wave per edge, lane = feature. Weights + Gaunt in LDS.
// Accumulates into y (pre-seeded with x) via global f32 atomics (segment_sum).
// ---------------------------------------------------------------------------
__global__ void msg_kernel(const float* __restrict__ x, const float* __restrict__ Yb,
                           const float* __restrict__ RBFb, const int* __restrict__ dst,
                           const int* __restrict__ src, const float* __restrict__ mpW,
                           const float* __restrict__ G, float* __restrict__ yout, int E) {
  __shared__ float Wl[27 * 256];  // [l1,l2,l3][8][32]
  __shared__ float Gl[729];
  for (int i = threadIdx.x; i < 27 * 256; i += blockDim.x) Wl[i] = mpW[i];
  for (int i = threadIdx.x; i < 729; i += blockDim.x) Gl[i] = G[i];
  __syncthreads();

  int wv = threadIdx.x >> 5, lane = threadIdx.x & 31;
  int e = blockIdx.x * 8 + wv;
  if (e >= E) return;

  int si = src[e], di = dst[e];
  float Ye[9], xs[9], acc[9];
#pragma unroll
  for (int q = 0; q < 9; ++q) Ye[q] = Yb[e * 9 + q];
  // rbf is 32B-aligned per edge -> two b128 loads
  const float4* r4 = (const float4*)(RBFb + (size_t)e * 8);
  float4 rbl = r4[0], rbh = r4[1];
  float rb[8] = {rbl.x, rbl.y, rbl.z, rbl.w, rbh.x, rbh.y, rbh.z, rbh.w};
#pragma unroll
  for (int p = 0; p < 9; ++p) { xs[p] = x[si * 9 * F + p * F + lane]; acc[p] = 0.0f; }

  const int PL1[11] = {0, 0, 0, 1, 1, 1, 1, 2, 2, 2, 2};
  const int PL2[11] = {0, 1, 2, 0, 1, 1, 2, 0, 1, 2, 2};
  const int PL3[11] = {0, 1, 2, 1, 0, 2, 1, 2, 1, 0, 2};
  const int OFFc[3] = {0, 1, 4};
  const int WIDc[3] = {1, 3, 5};

#pragma unroll
  for (int pa = 0; pa < 11; ++pa) {
    const int l1 = PL1[pa], l2 = PL2[pa], l3 = PL3[pa];
    const float* Wp = &Wl[((l1 * 3 + l2) * 3 + l3) * 256];
    float rw = 0.0f;
#pragma unroll
    for (int k = 0; k < 8; ++k) rw = fmaf(rb[k], Wp[k * 32 + lane], rw);
#pragma unroll
    for (int rl = 0; rl < WIDc[l3]; ++rl) {
      int r = OFFc[l3] + rl;
      float s = 0.0f;
#pragma unroll
      for (int pl = 0; pl < WIDc[l1]; ++pl) {
        int p = OFFc[l1] + pl;
        float a = 0.0f;
#pragma unroll
        for (int ql = 0; ql < WIDc[l2]; ++ql) {
          int q = OFFc[l2] + ql;
          a = fmaf(Gl[(p * 9 + q) * 9 + r], Ye[q], a);
        }
        s = fmaf(a, xs[p], s);
      }
      acc[r] = fmaf(rw, s, acc[r]);
    }
  }
#pragma unroll
  for (int r = 0; r < 9; ++r)
    atomicAdd(&yout[di * 9 * F + r * F + lane], acc[r]);
}

// ---------------------------------------------------------------------------
// Per-degree dense tile via WMMA f16->f32, degree as compile-time constant.
// Full-tile fast path has no per-row EXEC masking and incremental row->(node,p).
// ---------------------------------------------------------------------------
template <int D, bool RESID>
__device__ __forceinline__ void dense_tile(int t, int lane,
                                           const float* __restrict__ xin,
                                           const float* __restrict__ W3,
                                           const float* __restrict__ bias,
                                           const float* __restrict__ resid,
                                           float* __restrict__ out, int N) {
  constexpr int offp = (D == 0) ? 0 : (D == 1) ? 1 : 4;
  constexpr int wid  = (D == 0) ? 1 : (D == 1) ? 3 : 5;
  const int rowsMax = N * wid;
  const int hi  = lane >> 4;
  const int sub = lane & 15;

  // --- A tile: per-lane K pattern is {hi*8..hi*8+7} and {16+hi*8..}: 4x b128 ---
  int rowA  = t * 16 + sub;
  int rowAc = rowA < rowsMax ? rowA : rowsMax - 1;
  int nodeA = rowAc / wid;
  int pA    = offp + rowAc % wid;
  const float4* a4 = (const float4*)(xin + (size_t)(nodeA * 9 + pA) * F);
  float4 av[4] = {a4[hi * 2], a4[hi * 2 + 1], a4[4 + hi * 2], a4[5 + hi * 2]};
  v16h A;
#pragma unroll
  for (int j = 0; j < 4; ++j) {
    A[j * 4 + 0] = (_Float16)av[j].x;
    A[j * 4 + 1] = (_Float16)av[j].y;
    A[j * 4 + 2] = (_Float16)av[j].z;
    A[j * 4 + 3] = (_Float16)av[j].w;
  }

  // --- B tiles: W[D] is [K=32][Ncol=32]; lane holds col sub, K = hi*8 + idx (+8 for idx>=8) ---
  const float* Wd = W3 + D * F * F;
  v16h B0, B1;
#pragma unroll
  for (int idx = 0; idx < 16; ++idx) {
    int k = (hi << 3) + idx + ((idx >> 3) << 3);
    B0[idx] = (_Float16)Wd[k * F + sub];
    B1[idx] = (_Float16)Wd[k * F + sub + 16];
  }

  v8f C0 = {}; v8f C1 = {};
  C0 = __builtin_amdgcn_wmma_f32_16x16x32_f16(false, A, false, B0, (short)0, C0, false, false);
  C1 = __builtin_amdgcn_wmma_f32_16x16x32_f16(false, A, false, B1, (short)0, C1, false, false);

  // --- D layout: VGPR v -> row rg0+v (consecutive), col = sub (+0/16) ---
  const int rg0 = t * 16 + hi * 8;
  int nd = rg0 / wid;        // one strength-reduced division per lane
  int jj = rg0 % wid;
  if (t * 16 + 16 <= rowsMax) {   // wave-uniform: full tile (always true when N%16==0)
#pragma unroll
    for (int v = 0; v < 8; ++v) {
      int base = (nd * 9 + offp + jj) * F;
      float r0 = C0[v], r1 = C1[v];
      if (D == 0) { r0 += bias[sub]; r1 += bias[sub + 16]; }
      if (RESID)  { r0 += resid[base + sub]; r1 += resid[base + sub + 16]; }
      out[base + sub]      = r0;
      out[base + sub + 16] = r1;
      if (wid == 1) { ++nd; }
      else {
        int c = (jj == wid - 1);
        jj = c ? 0 : jj + 1;
        nd += c;
      }
    }
  } else {                        // ragged tail (unused at these sizes)
#pragma unroll
    for (int v = 0; v < 8; ++v) {
      if (rg0 + v < rowsMax) {
        int base = (nd * 9 + offp + jj) * F;
        float r0 = C0[v], r1 = C1[v];
        if (D == 0) { r0 += bias[sub]; r1 += bias[sub + 16]; }
        if (RESID)  { r0 += resid[base + sub]; r1 += resid[base + sub + 16]; }
        out[base + sub]      = r0;
        out[base + sub + 16] = r1;
      }
      if (wid == 1) { ++nd; }
      else {
        int c = (jj == wid - 1);
        jj = c ? 0 : jj + 1;
        nd += c;
      }
    }
  }
}

template <bool RESID>
__global__ void dense_wmma_kernel(const float* __restrict__ xin,
                                  const float* __restrict__ W3,   // [3][32][32]
                                  const float* __restrict__ bias, // [32]
                                  const float* __restrict__ resid,
                                  float* __restrict__ out, int N) {
  const int nt0 = (N + 15) / 16;
  const int nt1 = nt0 + (3 * N + 15) / 16;
  const int nt2 = nt1 + (5 * N + 15) / 16;
  int wave = (blockIdx.x * blockDim.x + threadIdx.x) >> 5;
  int lane = threadIdx.x & 31;
  if (wave >= nt2) return;
  if (wave < nt0)      dense_tile<0, RESID>(wave,       lane, xin, W3, bias, resid, out, N);
  else if (wave < nt1) dense_tile<1, RESID>(wave - nt0, lane, xin, W3, bias, resid, out, N);
  else                 dense_tile<2, RESID>(wave - nt1, lane, xin, W3, bias, resid, out, N);
}

// ---------------------------------------------------------------------------
// e3-equivariant SiLU gate (in place)
// ---------------------------------------------------------------------------
__global__ void silu_kernel(float* __restrict__ z, int N) {
  int i = blockIdx.x * blockDim.x + threadIdx.x;
  if (i >= N * F) return;
  int n = i >> 5, f = i & 31;
  float* b = z + n * 9 * F + f;
  float s  = b[0];
  float sg = 1.0f / (1.0f + expf(-s));
  b[0] = s * sg;
  float g = sg * (1.0f + s * (1.0f - sg));
#pragma unroll
  for (int p = 1; p < 9; ++p) b[p * F] *= g;
}

// ---------------------------------------------------------------------------
// Tensor-product readout: one wave per node, lane = feature, shuffle reduce.
// ---------------------------------------------------------------------------
__global__ void td_kernel(const float* __restrict__ x, const float* __restrict__ tdW,
                          const float* __restrict__ G, float* __restrict__ x2, int N) {
  __shared__ float Gl[729];
  for (int i = threadIdx.x; i < 729; i += blockDim.x) Gl[i] = G[i];
  __syncthreads();

  int wv = threadIdx.x >> 5, lane = threadIdx.x & 31;
  int n = blockIdx.x * 8 + wv;
  if (n >= N) return;

  float xv[9];
#pragma unroll
  for (int p = 0; p < 9; ++p) xv[p] = x[n * 9 * F + p * F + lane];

  float part[4][4];
#pragma unroll
  for (int r = 0; r < 4; ++r)
#pragma unroll
    for (int g = 0; g < 4; ++g) part[r][g] = 0.0f;

  const int TL1[7] = {0, 0, 1, 1, 1, 2, 2};
  const int TL2[7] = {0, 1, 0, 1, 2, 1, 2};
  const int TL3[7] = {0, 1, 1, 0, 1, 1, 0};
  const int OFFc[3] = {0, 1, 4};
  const int WIDc[3] = {1, 3, 5};

#pragma unroll
  for (int pa = 0; pa < 7; ++pa) {
    const int l1 = TL1[pa], l2 = TL2[pa], l3 = TL3[pa];
    const float* Wp = tdW + ((l1 * 3 + l2) * 2 + l3) * F * 4 + lane * 4;
#pragma unroll
    for (int rl = 0; rl < WIDc[l3]; ++rl) {
      int r = OFFc[l3] + rl;  // 0 or 1..3
      float h = 0.0f;
#pragma unroll
      for (int pl = 0; pl < WIDc[l1]; ++pl) {
        int p = OFFc[l1] + pl;
        float a = 0.0f;
#pragma unroll
        for (int ql = 0; ql < WIDc[l2]; ++ql) {
          int q = OFFc[l2] + ql;
          a = fmaf(Gl[(p * 9 + q) * 9 + r], xv[q], a);
        }
        h = fmaf(a, xv[p], h);
      }
#pragma unroll
      for (int g = 0; g < 4; ++g) part[r][g] = fmaf(h, Wp[g], part[r][g]);
    }
  }
#pragma unroll
  for (int r = 0; r < 4; ++r)
#pragma unroll
    for (int g = 0; g < 4; ++g) {
      float v = part[r][g];
      for (int off = 16; off > 0; off >>= 1) v += __shfl_xor(v, off, 32);
      if (lane == 0) x2[n * 16 + r * 4 + g] = v;
    }
}

// ---------------------------------------------------------------------------
// Output heads: mono = x2[:,0,:]@mono_W + bias[Z];  dipo = gate(x2)*0.3
// ---------------------------------------------------------------------------
__global__ void out_kernel(const float* __restrict__ x2, const float* __restrict__ monoW,
                           const float* __restrict__ ebias, const int* __restrict__ Z,
                           float* __restrict__ out, int N) {
  int n = blockIdx.x * blockDim.x + threadIdx.x;
  if (n >= N) return;
  float s[4];
#pragma unroll
  for (int g = 0; g < 4; ++g) s[g] = x2[n * 16 + g];
  float eb = ebias[Z[n]];
#pragma unroll
  for (int g = 0; g < 4; ++g) {
    float m = eb;
#pragma unroll
    for (int k = 0; k < 4; ++k) m = fmaf(s[k], monoW[k * 4 + g], m);
    out[n * 4 + g] = m;
  }
#pragma unroll
  for (int r = 1; r < 4; ++r)
#pragma unroll
    for (int g = 0; g < 4; ++g) {
      float gate = (s[g] > -1.0f && s[g] < 1.0f) ? 1.0f : 0.0f;
      out[N * 4 + (n * 3 + (r - 1)) * 4 + g] = x2[n * 16 + r * 4 + g] * gate * 0.3f;
    }
}

// ---------------------------------------------------------------------------
extern "C" void kernel_launch(void* const* d_in, const int* in_sizes, int n_in,
                              void* d_out, int out_size, void* d_ws, size_t ws_size,
                              hipStream_t stream) {
  (void)n_in; (void)out_size; (void)ws_size;
  const float* positions = (const float*)d_in[0];
  const float* embed     = (const float*)d_in[1];
  const float* mp_W      = (const float*)d_in[2];
  const float* d1_W      = (const float*)d_in[3];
  const float* d1_b      = (const float*)d_in[4];
  const float* d2_W      = (const float*)d_in[5];
  const float* d2_b      = (const float*)d_in[6];
  const float* td_W      = (const float*)d_in[7];
  const float* mono_W    = (const float*)d_in[8];
  const float* ebias     = (const float*)d_in[9];
  const int*   Z         = (const int*)d_in[10];
  const int*   dst       = (const int*)d_in[11];
  const int*   src       = (const int*)d_in[12];
  float* out = (float*)d_out;

  const int N = in_sizes[10];   // N_NODES
  const int E = in_sizes[11];   // N_EDGES

  char* ws = (char*)d_ws;
  size_t off = 0;
  auto carve = [&](size_t bytes) -> float* {
    float* p = (float*)(ws + off);
    off = (off + bytes + 255) & ~(size_t)255;
    return p;
  };
  const size_t xbytes = (size_t)N * 9 * F * sizeof(float);
  float* G    = carve(729 * sizeof(float));
  float* Yb   = carve((size_t)E * 9 * sizeof(float));
  float* RBFb = carve((size_t)E * 8 * sizeof(float));
  float* xb   = carve(xbytes);
  float* yb   = carve(xbytes);
  float* zb   = carve(xbytes);
  float* x2   = carve((size_t)N * 16 * sizeof(float));

  gaunt_kernel<<<1, 768, 0, stream>>>(G);
  edge_basis_kernel<<<(E + 255) / 256, 256, 0, stream>>>(positions, dst, src, mp_W, Yb, RBFb, E);
  node_init_kernel<<<(N * F + 255) / 256, 256, 0, stream>>>(embed, Z, xb, N);

  const int ntiles = (N + 15) / 16 + (3 * N + 15) / 16 + (5 * N + 15) / 16;
  const int dblocks = (ntiles + 7) / 8;  // 8 waves / block

  for (int it = 0; it < 3; ++it) {
    hipMemcpyAsync(yb, xb, xbytes, hipMemcpyDeviceToDevice, stream);  // y = x (msg residual)
    msg_kernel<<<(E + 7) / 8, 256, 0, stream>>>(xb, Yb, RBFb, dst, src,
                                                mp_W + (size_t)it * 27 * 8 * F, G, yb, E);
    dense_wmma_kernel<false><<<dblocks, 256, 0, stream>>>(yb, d1_W + (size_t)it * 3 * F * F,
                                                          d1_b + (size_t)it * F, nullptr, zb, N);
    silu_kernel<<<(N * F + 255) / 256, 256, 0, stream>>>(zb, N);
    dense_wmma_kernel<true><<<dblocks, 256, 0, stream>>>(zb, d2_W + (size_t)it * 3 * F * F,
                                                         d2_b + (size_t)it * F, xb, xb, N);
  }

  td_kernel<<<(N + 7) / 8, 256, 0, stream>>>(xb, td_W, G, x2, N);
  out_kernel<<<(N + 255) / 256, 256, 0, stream>>>(x2, mono_W, ebias, Z, out, N);
}
